// AFDEKM_Model_16587163697205
// MI455X (gfx1250) — matrix operand order repaired and checked by reading, not compiled, not run on hardware
//
#include <hip/hip_runtime.h>
#include <hip/hip_bf16.h>

typedef _Float16 v8h  __attribute__((ext_vector_type(8)));
typedef _Float16 v16h __attribute__((ext_vector_type(16)));
typedef float    v8f  __attribute__((ext_vector_type(8)));
typedef int      v4i  __attribute__((ext_vector_type(4)));

#define NROWS  65536
#define MTILE  64
#define SA     520   // f16 row stride for 64x512 LDS tiles (padded)
#define SP     40    // f16 row stride for [n][32] weight k-panels (padded)
#define SW2    264   // f16 row stride for 64x256 LDS tiles (padded)

#define GLOBAL_AS __attribute__((address_space(1)))
#define LDS_AS    __attribute__((address_space(3)))

// ---- CDNA5 async global->LDS copy (ASYNCcnt path), with safe fallback -----
__device__ __forceinline__ void async_copy16(const _Float16* g, _Float16* l) {
#if __has_builtin(__builtin_amdgcn_global_load_async_to_lds_b128)
  __builtin_amdgcn_global_load_async_to_lds_b128(
      (GLOBAL_AS v4i*)g, (LDS_AS v4i*)l, 0, 0);
#else
  *(uint4*)l = *(const uint4*)g;
#endif
}
__device__ __forceinline__ void async_join() {
#if __has_builtin(__builtin_amdgcn_global_load_async_to_lds_b128)
#if __has_builtin(__builtin_amdgcn_s_wait_asynccnt)
  __builtin_amdgcn_s_wait_asynccnt(0);
#else
  asm volatile("s_wait_asynccnt 0x0" ::: "memory");
#endif
#endif
}

// ---- WMMA fragment loaders (layouts per CDNA5 ISA 7.12.2) -----------------
// A 16x32 f16: lanes 0-15 hold row M=lane with K = kb+{0..7, 16..23};
//              lanes 16-31 hold row M=lane-16 with K = kb+{8..15, 24..31}.
__device__ __forceinline__ v16h load_a_frag(const _Float16* base, int stride,
                                            int r0, int kb, int lid) {
  const int row = r0 + (lid & 15);
  const int kh  = (lid >> 4) << 3;
  union { v16h v; v8h h[2]; } u;
  u.h[0] = *(const v8h*)(base + row * stride + kb + kh);
  u.h[1] = *(const v8h*)(base + row * stride + kb + kh + 16);
  return u.v;
}
// B 32x16 f16 (stored transposed, [n][k] row-major in LDS):
// lane n (0-15) holds column n, K = kb+{0..7, 16..23}; lane n+16 holds K = kb+{8..15, 24..31}.
__device__ __forceinline__ v16h load_b_frag(const _Float16* base, int stride,
                                            int n0, int kb, int lid) {
  const int n = n0 + (lid & 15);
  const int k = kb + ((lid >> 4) << 3);
  union { v16h v; v8h h[2]; } u;
  u.h[0] = *(const v8h*)(base + n * stride + k);
  u.h[1] = *(const v8h*)(base + n * stride + k + 16);
  return u.v;
}

__device__ __forceinline__ v8f wmma_f16(v16h a, v16h b, v8f c) {
  return __builtin_amdgcn_wmma_f32_16x16x32_f16(false, a, false, b,
                                                (short)0, c, false, false);
}

// Stage a 256n x 32k f16 weight panel (W^T layout, global row stride 512).
__device__ __forceinline__ void stage_panel(const _Float16* __restrict__ wT,
                                            int koff, _Float16* pnl, int tid) {
#pragma unroll
  for (int i = 0; i < 4; ++i) {
    int ch = tid + i * 256;
    int n = ch >> 2, j8 = (ch & 3) << 3;
    async_copy16(wT + (size_t)n * 512 + koff + j8, pnl + n * SP + j8);
  }
}

// 64x256 += 64x512 @ 512x256 with double-buffered async panel staging.
// One barrier + one asynccnt-join per k-step; DMA of panel ks+1 overlaps
// the 8 WMMAs of panel ks.
__device__ __forceinline__ void gemm512(const _Float16* atile,
                                        const _Float16* __restrict__ wT,
                                        _Float16* pnl0, _Float16* pnl1,
                                        v8f* acc, int tid, int lid,
                                        int rg, int cg) {
  stage_panel(wT, 0, pnl0, tid);
  for (int ks = 0; ks < 16; ++ks) {
    async_join();
    __syncthreads();
    _Float16* cur = (ks & 1) ? pnl1 : pnl0;
    _Float16* nxt = (ks & 1) ? pnl0 : pnl1;
    if (ks < 15) stage_panel(wT, (ks + 1) * 32, nxt, tid);
    v16h a = load_a_frag(atile, SA, rg * 16, ks * 32, lid);
#pragma unroll
    for (int t = 0; t < 8; ++t) {
      v16h b = load_b_frag(cur, SP, cg * 128 + t * 16, 0, lid);
      acc[t] = wmma_f16(a, b, acc[t]);
    }
  }
}

// ---- K0: convert + transpose weights to f16 [n][k] ------------------------
__global__ __launch_bounds__(256) void k_prep(
    const float* __restrict__ aw1, const float* __restrict__ aw2,
    const float* __restrict__ ew1, const float* __restrict__ ew2,
    _Float16* __restrict__ aw1T, _Float16* __restrict__ aw2T,
    _Float16* __restrict__ ew1T, _Float16* __restrict__ ew2T) {
  int id = blockIdx.x * 256 + threadIdx.x;
  if (id < 262144) {                       // aw1T: 512n x 512k
    int n = id >> 9, k = id & 511;
    aw1T[id] = (_Float16)aw1[k * 512 + n];
  } else if (id < 524288) {                // aw2T: 512n x 512k
    int l = id - 262144; int n = l >> 9, k = l & 511;
    aw2T[l] = (_Float16)aw2[k * 512 + n];
  } else if (id < 655360) {                // ew1T: 256n x 512k
    int l = id - 524288; int n = l >> 9, k = l & 511;
    ew1T[l] = (_Float16)ew1[k * 256 + n];
  } else if (id < 671744) {                // ew2T: 64n x 256k
    int l = id - 655360; int n = l >> 8, k = l & 255;
    ew2T[l] = (_Float16)ew2[k * 64 + n];
  }
}

// ---- K1: a = relu(x@aw1+ab1); logits = a@aw2+ab2 --------------------------
__global__ __launch_bounds__(256) void k_attn(
    const float* __restrict__ x,
    const float* __restrict__ ab1, const float* __restrict__ ab2,
    const _Float16* __restrict__ aw1T, const _Float16* __restrict__ aw2T,
    float* __restrict__ logits) {
  extern __shared__ _Float16 smem[];
  _Float16* xt   = smem;                       // 64 x SA
  _Float16* at   = smem + MTILE * SA;          // 64 x SA
  _Float16* pnl0 = smem + 2 * MTILE * SA;      // 256 x SP
  _Float16* pnl1 = pnl0 + 256 * SP;            // 256 x SP
  const int tid = threadIdx.x, lid = tid & 31, wv = tid >> 5;
  const int rg = wv >> 1, cg = wv & 1;
  const int rowBase = blockIdx.x * MTILE;
  const int rowOff = rg * 16 + ((lid >> 4) << 3);

  // x tile -> f16 LDS
#pragma unroll
  for (int i = 0; i < 32; ++i) {
    int e4 = tid + i * 256;
    int row = e4 >> 7, c4 = (e4 & 127) << 2;
    float4 v = *(const float4*)(x + (size_t)(rowBase + row) * 512 + c4);
    _Float16* d = xt + row * SA + c4;
    d[0] = (_Float16)v.x; d[1] = (_Float16)v.y;
    d[2] = (_Float16)v.z; d[3] = (_Float16)v.w;
  }

  // GEMM1: two 256-column halves, K=512
  for (int nh = 0; nh < 2; ++nh) {
    v8f acc[8];
#pragma unroll
    for (int t = 0; t < 8; ++t)
#pragma unroll
      for (int j = 0; j < 8; ++j) acc[t][j] = 0.f;
    gemm512(xt, aw1T + (size_t)(nh * 256) * 512, pnl0, pnl1, acc, tid, lid, rg, cg);
#pragma unroll
    for (int t = 0; t < 8; ++t) {
      int col = nh * 256 + cg * 128 + t * 16 + (lid & 15);
      float bias = ab1[col];
#pragma unroll
      for (int r = 0; r < 8; ++r)
        at[(rowOff + r) * SA + col] = (_Float16)fmaxf(acc[t][r] + bias, 0.f);
    }
    __syncthreads();
  }

  // GEMM2: logits -> global
  for (int nh = 0; nh < 2; ++nh) {
    v8f acc[8];
#pragma unroll
    for (int t = 0; t < 8; ++t)
#pragma unroll
      for (int j = 0; j < 8; ++j) acc[t][j] = 0.f;
    gemm512(at, aw2T + (size_t)(nh * 256) * 512, pnl0, pnl1, acc, tid, lid, rg, cg);
#pragma unroll
    for (int t = 0; t < 8; ++t) {
      int col = nh * 256 + cg * 128 + t * 16 + (lid & 15);
      float bias = ab2[col];
#pragma unroll
      for (int r = 0; r < 8; ++r)
        logits[(size_t)(rowBase + rowOff + r) * 512 + col] = acc[t][r] + bias;
    }
    __syncthreads();
  }
}

// ---- K2: softmax (in-place logits->attn), wf = attn*x, h1pre = wf@ew1+eb1 -
__global__ __launch_bounds__(256) void k_softmax_gemm3(
    const float* __restrict__ x, float* __restrict__ attn,
    const float* __restrict__ eb1, const _Float16* __restrict__ ew1T,
    float* __restrict__ h1pre) {
  extern __shared__ _Float16 smem[];
  _Float16* wft  = smem;                       // 64 x SA
  _Float16* pnl0 = smem + MTILE * SA;          // 256 x SP
  _Float16* pnl1 = pnl0 + 256 * SP;            // 256 x SP
  float* red = (float*)(pnl1 + 256 * SP);      // 256 f32
  const int tid = threadIdx.x, lid = tid & 31, wv = tid >> 5;
  const int rg = wv >> 1, cg = wv & 1;
  const int rowBase = blockIdx.x * MTILE;
  const int srow = tid >> 2, q = tid & 3;
  const size_t rix = (size_t)(rowBase + srow) * 512;

  float m = -3.4e38f;
  for (int j = 0; j < 128; ++j) m = fmaxf(m, attn[rix + q + 4 * j]);
  red[srow * 4 + q] = m;
  __syncthreads();
  m = fmaxf(fmaxf(red[srow * 4], red[srow * 4 + 1]),
            fmaxf(red[srow * 4 + 2], red[srow * 4 + 3]));
  __syncthreads();
  float s = 0.f;
  for (int j = 0; j < 128; ++j) s += __expf(attn[rix + q + 4 * j] - m);
  red[srow * 4 + q] = s;
  __syncthreads();
  s = (red[srow * 4] + red[srow * 4 + 1]) + (red[srow * 4 + 2] + red[srow * 4 + 3]);
  float rs = 1.f / s;
  for (int j = 0; j < 128; ++j) {
    int c = q + 4 * j;
    float p = __expf(attn[rix + c] - m) * rs;
    wft[srow * SA + c] = (_Float16)(p * x[rix + c]);
    attn[rix + c] = p;
  }
  __syncthreads();

  v8f acc[8];
#pragma unroll
  for (int t = 0; t < 8; ++t)
#pragma unroll
    for (int j = 0; j < 8; ++j) acc[t][j] = 0.f;
  gemm512(wft, ew1T, pnl0, pnl1, acc, tid, lid, rg, cg);
  const int rowOff = rg * 16 + ((lid >> 4) << 3);
#pragma unroll
  for (int t = 0; t < 8; ++t) {
    int col = cg * 128 + t * 16 + (lid & 15);
    float bias = eb1[col];
#pragma unroll
    for (int r = 0; r < 8; ++r)
      h1pre[(size_t)(rowBase + rowOff + r) * 256 + col] = acc[t][r] + bias;
  }
}

// ---- BN1 stats: deterministic two-stage reduction -------------------------
__global__ __launch_bounds__(256) void k_bn1_part(
    const float* __restrict__ h1pre, float* __restrict__ ps, float* __restrict__ ps2) {
  int b = blockIdx.x, t = threadIdx.x;
  float s = 0.f, s2 = 0.f;
  size_t base = (size_t)b * 256 * 256 + t;
  for (int r = 0; r < 256; ++r) {
    float v = h1pre[base + (size_t)r * 256]; s += v; s2 += v * v;
  }
  ps[b * 256 + t] = s; ps2[b * 256 + t] = s2;
}
__global__ __launch_bounds__(256) void k_bn1_fin(
    const float* __restrict__ ps, const float* __restrict__ ps2,
    const float* __restrict__ g, const float* __restrict__ bt,
    float* __restrict__ scale, float* __restrict__ shift) {
  int t = threadIdx.x;
  float s = 0.f, s2 = 0.f;
  for (int b = 0; b < 256; ++b) { s += ps[b * 256 + t]; s2 += ps2[b * 256 + t]; }
  float mu = s * (1.f / 65536.f);
  float var = s2 * (1.f / 65536.f) - mu * mu;
  float sc = g[t] * rsqrtf(var + 1e-5f);
  scale[t] = sc; shift[t] = bt[t] - mu * sc;
}

// ---- K4: h1 = relu(bn1(h1pre)); h2pre = h1@ew2+eb2 ------------------------
__global__ __launch_bounds__(256) void k_enc2(
    const float* __restrict__ h1pre,
    const float* __restrict__ scale1, const float* __restrict__ shift1,
    const float* __restrict__ eb2, const _Float16* __restrict__ ew2T,
    float* __restrict__ h2pre) {
  extern __shared__ _Float16 smem[];
  _Float16* h1t = smem;                    // 64 x SW2
  _Float16* w2t = smem + MTILE * SW2;      // 64 x SW2 (full ew2T)
  const int tid = threadIdx.x, lid = tid & 31, wv = tid >> 5;
  const int rg = wv >> 1, cg = wv & 1;
  const int rowBase = blockIdx.x * MTILE;
#pragma unroll
  for (int i = 0; i < 8; ++i) {            // async stage ew2T (64n x 256k f16)
    int ch = tid + i * 256;
    int n = ch >> 5, j8 = (ch & 31) << 3;
    async_copy16(ew2T + n * 256 + j8, w2t + n * SW2 + j8);
  }
  for (int i = 0; i < 64; ++i) {           // h1 tile with bn+relu (overlaps DMA)
    int e = tid + i * 256;
    int row = e >> 8, c = e & 255;
    float v = h1pre[(size_t)(rowBase + row) * 256 + c] * scale1[c] + shift1[c];
    h1t[row * SW2 + c] = (_Float16)fmaxf(v, 0.f);
  }
  async_join();
  __syncthreads();
  v8f acc[2];
#pragma unroll
  for (int t = 0; t < 2; ++t)
#pragma unroll
    for (int j = 0; j < 8; ++j) acc[t][j] = 0.f;
  for (int ks = 0; ks < 8; ++ks) {
    v16h a = load_a_frag(h1t, SW2, rg * 16, ks * 32, lid);
#pragma unroll
    for (int t = 0; t < 2; ++t) {
      v16h b = load_b_frag(w2t, SW2, cg * 32 + t * 16, ks * 32, lid);
      acc[t] = wmma_f16(a, b, acc[t]);
    }
  }
  const int rowOff = rg * 16 + ((lid >> 4) << 3);
#pragma unroll
  for (int t = 0; t < 2; ++t) {
    int col = cg * 32 + t * 16 + (lid & 15);
    float bias = eb2[col];
#pragma unroll
    for (int r = 0; r < 8; ++r)
      h2pre[(size_t)(rowBase + rowOff + r) * 64 + col] = acc[t][r] + bias;
  }
}

// ---- BN2 stats ------------------------------------------------------------
__global__ __launch_bounds__(64) void k_bn2_part(
    const float* __restrict__ h2pre, float* __restrict__ ps, float* __restrict__ ps2) {
  int b = blockIdx.x, t = threadIdx.x;
  float s = 0.f, s2 = 0.f;
  size_t base = (size_t)b * 256 * 64 + t;
  for (int r = 0; r < 256; ++r) {
    float v = h2pre[base + (size_t)r * 64]; s += v; s2 += v * v;
  }
  ps[b * 64 + t] = s; ps2[b * 64 + t] = s2;
}
__global__ __launch_bounds__(64) void k_bn2_fin(
    const float* __restrict__ ps, const float* __restrict__ ps2,
    const float* __restrict__ g, const float* __restrict__ bt,
    float* __restrict__ scale, float* __restrict__ shift) {
  int t = threadIdx.x;
  float s = 0.f, s2 = 0.f;
  for (int b = 0; b < 256; ++b) { s += ps[b * 64 + t]; s2 += ps2[b * 64 + t]; }
  float mu = s * (1.f / 65536.f);
  float var = s2 * (1.f / 65536.f) - mu * mu;
  float sc = g[t] * rsqrtf(var + 1e-5f);
  scale[t] = sc; shift[t] = bt[t] - mu * sc;
}

// ---- K6: enc = tanh(bn2(h2pre)); fuzzy membership -------------------------
__global__ __launch_bounds__(256) void k_member(
    const float* __restrict__ h2pre,
    const float* __restrict__ scale2, const float* __restrict__ shift2,
    const float* __restrict__ centers,
    float* __restrict__ enc_out, float* __restrict__ mem_out) {
  __shared__ float et[64 * 65];
  __shared__ float cs[64 * 65];
  __shared__ float psum[64 * 4];
  const int tid = threadIdx.x;
  const int rowBase = blockIdx.x * MTILE;
#pragma unroll
  for (int i = 0; i < 16; ++i) {
    int e = tid + i * 256;
    int k = e >> 6, d = e & 63;
    cs[k * 65 + d] = centers[e];
  }
  for (int i = 0; i < 16; ++i) {
    int e = tid + i * 256;
    int row = e >> 6, d = e & 63;
    float v = tanhf(h2pre[(size_t)(rowBase + row) * 64 + d] * scale2[d] + shift2[d]);
    et[row * 65 + d] = v;
    enc_out[(size_t)(rowBase + row) * 64 + d] = v;
  }
  __syncthreads();
  const int row = tid >> 2, kg = (tid & 3) << 4;
  float inv[16];
  float tot = 0.f;
#pragma unroll
  for (int kk = 0; kk < 16; ++kk) {
    int k = kg + kk;
    float d2 = 0.f;
    for (int d = 0; d < 64; ++d) {
      float df = et[row * 65 + d] - cs[k * 65 + d];
      d2 = fmaf(df, df, d2);
    }
    // dist = max(sqrt(max(d2,0)),1e-10); power=2/(2-1)=2 -> inv = 1/max(d2,1e-20)
    float iv = 1.f / fmaxf(d2, 1e-20f);
    inv[kk] = iv; tot += iv;
  }
  psum[row * 4 + (tid & 3)] = tot;
  __syncthreads();
  float T = (psum[row * 4] + psum[row * 4 + 1]) + (psum[row * 4 + 2] + psum[row * 4 + 3]);
  float rT = 1.f / T;
#pragma unroll
  for (int kk = 0; kk < 16; ++kk)
    mem_out[(size_t)(rowBase + row) * 64 + kg + kk] = inv[kk] * rT;
}

extern "C" void kernel_launch(void* const* d_in, const int* in_sizes, int n_in,
                              void* d_out, int out_size, void* d_ws, size_t ws_size,
                              hipStream_t stream) {
  (void)in_sizes; (void)n_in; (void)out_size; (void)ws_size;
  const float* x       = (const float*)d_in[0];
  const float* aw1     = (const float*)d_in[1];
  const float* ab1     = (const float*)d_in[2];
  const float* aw2     = (const float*)d_in[3];
  const float* ab2     = (const float*)d_in[4];
  const float* ew1     = (const float*)d_in[5];
  const float* eb1     = (const float*)d_in[6];
  const float* g1      = (const float*)d_in[7];
  const float* b1      = (const float*)d_in[8];
  const float* ew2     = (const float*)d_in[9];
  const float* eb2     = (const float*)d_in[10];
  const float* g2      = (const float*)d_in[11];
  const float* b2      = (const float*)d_in[12];
  const float* centers = (const float*)d_in[13];

  char* w = (char*)d_ws;
  _Float16* aw1T = (_Float16*)(w + 0);
  _Float16* aw2T = (_Float16*)(w + 524288);
  _Float16* ew1T = (_Float16*)(w + 1048576);
  _Float16* ew2T = (_Float16*)(w + 1310720);
  float* h1pre   = (float*)(w + 1343488);      // 64 MB
  float* h2pre   = (float*)(w + 68452352);     // 16 MB
  float* bn1s    = (float*)(w + 85229568);
  float* bn1s2   = (float*)(w + 85491712);
  float* bn2s    = (float*)(w + 85753856);
  float* bn2s2   = (float*)(w + 85819392);
  float* sc1     = (float*)(w + 85884928);
  float* sh1     = sc1 + 256;
  float* sc2     = (float*)(w + 85886976);
  float* sh2     = sc2 + 64;

  float* attn = (float*)d_out;
  float* enc  = attn + (size_t)NROWS * 512;
  float* memb = enc + (size_t)NROWS * 64;

  const int nblk = NROWS / MTILE;  // 1024
  const size_t sm1 = (size_t)(2 * MTILE * SA + 2 * 256 * SP) * sizeof(_Float16);        // 174080
  const size_t sm2 = (size_t)(MTILE * SA + 2 * 256 * SP) * sizeof(_Float16) + 256 * 4;  // 108544
  const size_t sm4 = (size_t)(2 * MTILE * SW2) * sizeof(_Float16);                      //  67584

  k_prep<<<2624, 256, 0, stream>>>(aw1, aw2, ew1, ew2, aw1T, aw2T, ew1T, ew2T);
  k_attn<<<nblk, 256, sm1, stream>>>(x, ab1, ab2, aw1T, aw2T, attn);
  k_softmax_gemm3<<<nblk, 256, sm2, stream>>>(x, attn, eb1, ew1T, h1pre);
  k_bn1_part<<<256, 256, 0, stream>>>(h1pre, bn1s, bn1s2);
  k_bn1_fin<<<1, 256, 0, stream>>>(bn1s, bn1s2, g1, b1, sc1, sh1);
  k_enc2<<<nblk, 256, sm4, stream>>>(h1pre, sc1, sh1, eb2, ew2T, h2pre);
  k_bn2_part<<<256, 64, 0, stream>>>(h2pre, bn2s, bn2s2);
  k_bn2_fin<<<1, 64, 0, stream>>>(bn2s, bn2s2, g2, b2, sc2, sh2);
  k_member<<<nblk, 256, 0, stream>>>(h2pre, sc2, sh2, centers, enc, memb);
}
